// TransformerModel_78142634983899
// MI455X (gfx1250) — compile-verified, matrix operand order checked by reading, not run on hardware
//
#include <hip/hip_runtime.h>
#include <math.h>

// ---------------------------------------------------------------------------
// Model constants (match reference)
// ---------------------------------------------------------------------------
#define B_  2
#define S_  2048
#define D_  1024
#define H_  16
#define HD_ 64
#define FF_ 4096
#define V_  32000
#define L_  4
#define SCALE_ 0.125f   // HD^-0.5

typedef __bf16 bf16;
typedef __attribute__((ext_vector_type(16))) __bf16 bf16x16;
typedef __attribute__((ext_vector_type(8)))  __bf16 bf16x8;
typedef __attribute__((ext_vector_type(8)))  float  f32x8;
typedef unsigned int u32;
typedef __attribute__((ext_vector_type(4))) unsigned int u32x4;
typedef __attribute__((ext_vector_type(8))) int i32x8;
typedef __attribute__((ext_vector_type(4))) int i32x4;

// ---------------------------------------------------------------------------
// Utility: block-wide (256 threads, 8 waves of 32) sum reduction
// ---------------------------------------------------------------------------
__device__ __forceinline__ float blockSum256(float v, float* red8) {
#pragma unroll
    for (int m = 16; m >= 1; m >>= 1) v += __shfl_xor(v, m, 32);
    __syncthreads();                       // protect red8 across calls
    if ((threadIdx.x & 31) == 0) red8[threadIdx.x >> 5] = v;
    __syncthreads();
    float t = 0.f;
#pragma unroll
    for (int i = 0; i < 8; ++i) t += red8[i];
    return t;
}

// ---------------------------------------------------------------------------
// Embedding gather + LayerNorm -> fp32 residual stream. One row per block.
// ---------------------------------------------------------------------------
__global__ __launch_bounds__(256)
void embed_ln_kernel(const int* __restrict__ tokens, const float* __restrict__ emb,
                     const float* __restrict__ w, const float* __restrict__ bvec,
                     float* __restrict__ X) {
    __shared__ float red8[8];
    const int row = blockIdx.x;
    const int tok = tokens[row];
    const float* er = emb + (size_t)tok * D_;
    float loc[4]; float s = 0.f;
#pragma unroll
    for (int i = 0; i < 4; ++i) { loc[i] = er[threadIdx.x + 256 * i]; s += loc[i]; }
    const float mu = blockSum256(s, red8) * (1.f / D_);
    float vs = 0.f;
#pragma unroll
    for (int i = 0; i < 4; ++i) { float d = loc[i] - mu; vs += d * d; }
    const float var = blockSum256(vs, red8) * (1.f / D_);
    const float r = rsqrtf(var + 1e-5f);
#pragma unroll
    for (int i = 0; i < 4; ++i) {
        int c = threadIdx.x + 256 * i;
        X[(size_t)row * D_ + c] = (loc[i] - mu) * r * w[c] + bvec[c];
    }
}

// ---------------------------------------------------------------------------
// LayerNorm of fp32 rows -> bf16 output (GEMM operand). One row per block.
// ---------------------------------------------------------------------------
__global__ __launch_bounds__(256)
void ln_bf16_kernel(const float* __restrict__ X, const float* __restrict__ w,
                    const float* __restrict__ bvec, bf16* __restrict__ out) {
    __shared__ float red8[8];
    const int row = blockIdx.x;
    const float* xr = X + (size_t)row * D_;
    float loc[4]; float s = 0.f;
#pragma unroll
    for (int i = 0; i < 4; ++i) { loc[i] = xr[threadIdx.x + 256 * i]; s += loc[i]; }
    const float mu = blockSum256(s, red8) * (1.f / D_);
    float vs = 0.f;
#pragma unroll
    for (int i = 0; i < 4; ++i) { float d = loc[i] - mu; vs += d * d; }
    const float var = blockSum256(vs, red8) * (1.f / D_);
    const float r = rsqrtf(var + 1e-5f);
#pragma unroll
    for (int i = 0; i < 4; ++i) {
        int c = threadIdx.x + 256 * i;
        out[(size_t)row * D_ + c] = (bf16)((loc[i] - mu) * r * w[c] + bvec[c]);
    }
}

// ---------------------------------------------------------------------------
// Tiled GEMM: C[M,N] = A[M,K] (bf16) x W[K,N] (fp32, converted in staging)
//             + bias (+ residual), f32 WMMA accumulation.
// Block tile 128x128, BK=64, 256 threads = 8 waves (2 M x 4 N); each wave owns
// 64x32 = 4x2 accumulators and runs 16 v_wmma per stage.
//
// Grid: x = M tiles (fast), y = N tiles  -> all 32 M-blocks of one weight
// panel launch back-to-back so the K x 128 fp32 panel stays hot in L2 and
// weights stream from HBM exactly once.
//
// A-tile staging uses the Tensor Data Mover: wave 0 issues
// tensor_load_to_lds with a D# describing a 128x64-elem bf16 tile with
// row stride K, LDS-padded 4 DWORDs every 32 DWORDs (=> LDS stride 72 elems),
// then waits TENSORcnt==0 and the workgroup barrier publishes the tile.
// W-tile is staged transposed (scalar scatter) with fused fp32->bf16 convert.
//
// A-fragment layout (ISA 7.12.2, 16-bit A 16x32): lane row = lane&15,
// K runs [half*8,+8) and [16+half*8,+8) -> two contiguous 16B LDS loads.
// C/D layout: row = i + 8*(lane>>4), col = lane&15.
// ---------------------------------------------------------------------------
template <int OUT_BF16>
__global__ __launch_bounds__(256)
void gemm_bf16_kernel(const bf16* __restrict__ A, const float* __restrict__ W,
                      const float* __restrict__ bias, const float* __restrict__ res,
                      void* __restrict__ out, int M, int N, int K) {
    constexpr int BM = 128, BN = 128, BK = 64, LDT = BK + 8;  // LDS stride 72
    __shared__ bf16 lA[BM * LDT];
    __shared__ bf16 lB[BN * LDT];

    const int tid  = threadIdx.x;
    const int wv   = tid >> 5;
    const int lane = tid & 31;
    const int half = lane >> 4;
    const int l16  = lane & 15;
    const int wm   = wv >> 2;          // 0..1
    const int wn   = wv & 3;           // 0..3
    const int rowBase = blockIdx.x * BM;   // M is the fast grid dim
    const int colBase = blockIdx.y * BN;

    f32x8 acc[4][2];
#pragma unroll
    for (int mt = 0; mt < 4; ++mt)
#pragma unroll
        for (int nt = 0; nt < 2; ++nt)
#pragma unroll
            for (int i = 0; i < 8; ++i) acc[mt][nt][i] = 0.f;

    // --- TDM descriptor group 1 (loop-invariant): data_size=2B, pad enabled,
    //     pad_interval=32 DWORDs (code 4), pad_amount=4 DWORDs (code 3),
    //     tensor_dim0=K, tensor_dim1=M, tile_dim0=64, tile_dim1=128.
    const u32 ldsA = (u32)(uintptr_t)(&lA[0]);
    const unsigned long long gbase =
        (unsigned long long)(uintptr_t)A + ((unsigned long long)rowBase * K) * 2ull;
    i32x8 g1;
    g1[0] = (int)((1u << 16) | (1u << 20) | (4u << 22) | (3u << 25));
    g1[1] = (int)(((u32)K & 0xFFFFu) << 16);
    g1[2] = (int)((((u32)K >> 16) & 0xFFFFu) | (((u32)M & 0xFFFFu) << 16));
    g1[3] = (int)((((u32)M >> 16) & 0xFFFFu) | (64u << 16));
    g1[4] = 128;                       // tile_dim1 = 128, tile_dim2 = 0
    g1[5] = (int)(u32)K;               // tensor_dim0_stride low 32
    g1[6] = 0;                         // stride hi / dim1_stride (unused, 2D)
    g1[7] = 0;
    i32x4 gz; gz[0] = 0; gz[1] = 0; gz[2] = 0; gz[3] = 0;
    i32x8 gz8;
#pragma unroll
    for (int i = 0; i < 8; ++i) gz8[i] = 0;

    for (int k0 = 0; k0 < K; k0 += BK) {
        // ---- stage A tile via Tensor Data Mover (wave 0 issues; EXEC ignored)
        if (wv == 0) {
            unsigned long long ga = gbase + (unsigned long long)k0 * 2ull;
            u32x4 g0;
            g0[0] = 1u;                              // count=1, user descriptor
            g0[1] = ldsA;                            // LDS byte address of lA
            g0[2] = (u32)ga;                         // global_addr[31:0]
            g0[3] = (u32)(ga >> 32) | (2u << 30);    // global_addr[56:32] | type=2
            __builtin_amdgcn_tensor_load_to_lds(g0, g1, gz, gz, gz8, 0);
            __builtin_amdgcn_s_wait_tensorcnt(0);
        }
        // ---- stage W tile transposed with fused fp32->bf16 convert
        for (int c = tid; c < 2048; c += 256) {
            int kr = c >> 5, nc = (c & 31) * 4;
            const float4 wv4 = *(const float4*)(W + (size_t)(k0 + kr) * N + colBase + nc);
            lB[(nc + 0) * LDT + kr] = (bf16)wv4.x;
            lB[(nc + 1) * LDT + kr] = (bf16)wv4.y;
            lB[(nc + 2) * LDT + kr] = (bf16)wv4.z;
            lB[(nc + 3) * LDT + kr] = (bf16)wv4.w;
        }
        __syncthreads();

#pragma unroll
        for (int f = 0; f < 2; ++f) {              // two K=32 WMMA steps
            bf16x16 afr[4];
#pragma unroll
            for (int mt = 0; mt < 4; ++mt) {
                int r = wm * 64 + mt * 16 + l16;
                union { bf16x16 v; bf16x8 h[2]; } u;
                u.h[0] = *(const bf16x8*)(&lA[r * LDT + f * 32 + half * 8]);
                u.h[1] = *(const bf16x8*)(&lA[r * LDT + f * 32 + 16 + half * 8]);
                afr[mt] = u.v;
            }
            bf16x16 bfr[2];
#pragma unroll
            for (int nt = 0; nt < 2; ++nt) {
                int cc = wn * 32 + nt * 16 + l16;
                union { bf16x16 v; bf16x8 h[2]; } u;
                u.h[0] = *(const bf16x8*)(&lB[cc * LDT + f * 32 + half * 8]);
                u.h[1] = *(const bf16x8*)(&lB[cc * LDT + f * 32 + 16 + half * 8]);
                bfr[nt] = u.v;
            }
#pragma unroll
            for (int mt = 0; mt < 4; ++mt)
#pragma unroll
                for (int nt = 0; nt < 2; ++nt)
                    acc[mt][nt] = __builtin_amdgcn_wmma_f32_16x16x32_bf16(
                        false, afr[mt], false, bfr[nt], (short)0, acc[mt][nt],
                        false, false);
        }
        __syncthreads();
    }

    // ---- epilogue: bias + residual, store fp32 or bf16
#pragma unroll
    for (int mt = 0; mt < 4; ++mt) {
#pragma unroll
        for (int nt = 0; nt < 2; ++nt) {
#pragma unroll
            for (int i = 0; i < 8; ++i) {
                int r  = rowBase + wm * 64 + mt * 16 + half * 8 + i;
                int cc = colBase + wn * 32 + nt * 16 + l16;
                float v = acc[mt][nt][i];
                if (bias) v += bias[cc];
                size_t idx = (size_t)r * N + cc;
                if (res) v += res[idx];
                if (OUT_BF16) ((bf16*)out)[idx] = (bf16)v;
                else          ((float*)out)[idx] = v;
            }
        }
    }
}

// ---------------------------------------------------------------------------
// Flash attention (non-causal, matches reference full softmax).
// Grid: B*H*(S/64) blocks; block = 128 threads = 4 waves; each wave owns a
// 16-row query tile. Q in registers; K-tile and V^T-tile in LDS; online
// softmax per row (16-lane shuffle reductions); P restaged via LDS as bf16.
// Q/K/V/O are [B,S,D] bf16 with heads at stride HD_.
// ---------------------------------------------------------------------------
__global__ __launch_bounds__(128)
void flash_attn_kernel(const bf16* __restrict__ Q, const bf16* __restrict__ Kv,
                       const bf16* __restrict__ Vv, bf16* __restrict__ O) {
    constexpr int LKT = HD_ + 8;                 // padded LDS stride (72)
    __shared__ bf16 lK [64 * LKT];               // [key][hd]
    __shared__ bf16 lVt[64 * LKT];               // [hd][key]
    __shared__ bf16 lP [4 * 16 * LKT];           // per-wave P staging

    const int tid  = threadIdx.x;
    const int wv   = tid >> 5;
    const int lane = tid & 31;
    const int half = lane >> 4;
    const int l16  = lane & 15;
    const int qt = blockIdx.x & 31;              // S/64 = 32 query tiles
    const int bh = blockIdx.x >> 5;
    const int b  = bh / H_, h = bh % H_;
    const size_t headOff = (size_t)h * HD_;
    const int qrow0 = qt * 64 + wv * 16;

    // Q fragments (K=64 -> two 16x32 A-fragments), straight from global
    bf16x16 qf[2];
    {
        const bf16* qr = Q + ((size_t)(b * S_ + qrow0 + l16)) * D_ + headOff;
#pragma unroll
        for (int f = 0; f < 2; ++f) {
            union { bf16x16 v; bf16x8 hh[2]; } u;
            u.hh[0] = *(const bf16x8*)(qr + f * 32 + half * 8);
            u.hh[1] = *(const bf16x8*)(qr + f * 32 + 16 + half * 8);
            qf[f] = u.v;
        }
    }

    float mrow[8], lrow[8];
    f32x8 oacc[4];
#pragma unroll
    for (int i = 0; i < 8; ++i) {
        mrow[i] = -3.0e38f; lrow[i] = 0.f;
#pragma unroll
        for (int nt = 0; nt < 4; ++nt) oacc[nt][i] = 0.f;
    }

    for (int kt = 0; kt < S_ / 64; ++kt) {
        // cooperative stage: K tile row-major, V tile transposed
        for (int c = tid; c < 512; c += 128) {
            int kr = c >> 3, hd8 = (c & 7) * 8;
            const size_t g = ((size_t)(b * S_ + kt * 64 + kr)) * D_ + headOff + hd8;
            *(uint4*)(&lK[kr * LKT + hd8]) = *(const uint4*)(Kv + g);
            bf16x8 vv = *(const bf16x8*)(Vv + g);
#pragma unroll
            for (int j = 0; j < 8; ++j) lVt[(hd8 + j) * LKT + kr] = vv[j];
        }
        __syncthreads();

        // S = Q * K^T  (4 key-subtiles of 16)
        f32x8 sc[4];
#pragma unroll
        for (int nt = 0; nt < 4; ++nt) {
            f32x8 z;
#pragma unroll
            for (int i = 0; i < 8; ++i) z[i] = 0.f;
            int kc = nt * 16 + l16;
#pragma unroll
            for (int f = 0; f < 2; ++f) {
                union { bf16x16 v; bf16x8 hh[2]; } u;
                u.hh[0] = *(const bf16x8*)(&lK[kc * LKT + f * 32 + half * 8]);
                u.hh[1] = *(const bf16x8*)(&lK[kc * LKT + f * 32 + 16 + half * 8]);
                z = __builtin_amdgcn_wmma_f32_16x16x32_bf16(
                        false, qf[f], false, u.v, (short)0, z, false, false);
            }
            sc[nt] = z;
        }

        // online softmax per row (row lives in one 16-lane half)
#pragma unroll
        for (int i = 0; i < 8; ++i) {
            float mx = -3.0e38f;
#pragma unroll
            for (int nt = 0; nt < 4; ++nt) {
                float v = sc[nt][i] * SCALE_; sc[nt][i] = v; mx = fmaxf(mx, v);
            }
#pragma unroll
            for (int m = 8; m >= 1; m >>= 1) mx = fmaxf(mx, __shfl_xor(mx, m, 32));
            float mn   = fmaxf(mrow[i], mx);
            float corr = __expf(mrow[i] - mn);
            float ssum = 0.f;
#pragma unroll
            for (int nt = 0; nt < 4; ++nt) {
                float p = __expf(sc[nt][i] - mn); sc[nt][i] = p; ssum += p;
            }
#pragma unroll
            for (int m = 8; m >= 1; m >>= 1) ssum += __shfl_xor(ssum, m, 32);
            lrow[i] = lrow[i] * corr + ssum;
            mrow[i] = mn;
#pragma unroll
            for (int nt = 0; nt < 4; ++nt) oacc[nt][i] *= corr;
        }

        // stage P (C-layout) to LDS as bf16 for A-fragment reload
        bf16* lp = &lP[wv * 16 * LKT];
#pragma unroll
        for (int nt = 0; nt < 4; ++nt)
#pragma unroll
            for (int i = 0; i < 8; ++i)
                lp[(half * 8 + i) * LKT + nt * 16 + l16] = (bf16)sc[nt][i];
        __syncthreads();

        // O += P * V
        bf16x16 pf[2];
#pragma unroll
        for (int f = 0; f < 2; ++f) {
            union { bf16x16 v; bf16x8 hh[2]; } u;
            u.hh[0] = *(const bf16x8*)(&lp[l16 * LKT + f * 32 + half * 8]);
            u.hh[1] = *(const bf16x8*)(&lp[l16 * LKT + f * 32 + 16 + half * 8]);
            pf[f] = u.v;
        }
#pragma unroll
        for (int nt = 0; nt < 4; ++nt) {
            int hc = nt * 16 + l16;
#pragma unroll
            for (int f = 0; f < 2; ++f) {
                union { bf16x16 v; bf16x8 hh[2]; } u;
                u.hh[0] = *(const bf16x8*)(&lVt[hc * LKT + f * 32 + half * 8]);
                u.hh[1] = *(const bf16x8*)(&lVt[hc * LKT + f * 32 + 16 + half * 8]);
                oacc[nt] = __builtin_amdgcn_wmma_f32_16x16x32_bf16(
                               false, pf[f], false, u.v, (short)0, oacc[nt],
                               false, false);
            }
        }
        __syncthreads();
    }

    // normalize and write O
#pragma unroll
    for (int nt = 0; nt < 4; ++nt)
#pragma unroll
        for (int i = 0; i < 8; ++i) {
            int s  = qrow0 + half * 8 + i;
            int hd = nt * 16 + l16;
            float v = oacc[nt][i] / lrow[i];
            O[((size_t)(b * S_ + s)) * D_ + headOff + hd] = (bf16)v;
        }
}

// ---------------------------------------------------------------------------
// GEGLU: ff = gelu_exact(gate) * val, bf16 in / bf16 out
// ---------------------------------------------------------------------------
__global__ void geglu_kernel(const bf16* __restrict__ gv, bf16* __restrict__ ff,
                             int M) {
    const size_t n = (size_t)M * FF_;
    const size_t stride = (size_t)gridDim.x * blockDim.x;
    for (size_t i = (size_t)blockIdx.x * blockDim.x + threadIdx.x; i < n; i += stride) {
        size_t r = i / FF_, c = i % FF_;
        float g = (float)gv[r * (2 * FF_) + c];
        float v = (float)gv[r * (2 * FF_) + FF_ + c];
        float ge = 0.5f * g * (1.f + erff(g * 0.70710678118654752f));
        ff[i] = (bf16)(ge * v);
    }
}

// ---------------------------------------------------------------------------
// Host-side orchestration
// ---------------------------------------------------------------------------
extern "C" void kernel_launch(void* const* d_in, const int* in_sizes, int n_in,
                              void* d_out, int out_size, void* d_ws, size_t ws_size,
                              hipStream_t stream) {
    (void)in_sizes; (void)n_in; (void)out_size; (void)ws_size;

    const int*   tokens  = (const int*)  d_in[0];
    const float* tok_emb = (const float*)d_in[1];
    const float* ln_e_w  = (const float*)d_in[2];
    const float* ln_e_b  = (const float*)d_in[3];
    const float* ln1_w   = (const float*)d_in[4];
    const float* ln1_b   = (const float*)d_in[5];
    const float* wq      = (const float*)d_in[6];
    const float* bq      = (const float*)d_in[7];
    const float* wk      = (const float*)d_in[8];
    const float* bk      = (const float*)d_in[9];
    const float* wvw     = (const float*)d_in[10];
    const float* bv      = (const float*)d_in[11];
    const float* wo      = (const float*)d_in[12];
    const float* bo      = (const float*)d_in[13];
    const float* ln2_w   = (const float*)d_in[14];
    const float* ln2_b   = (const float*)d_in[15];
    const float* w1      = (const float*)d_in[16];
    const float* b1      = (const float*)d_in[17];
    const float* w2      = (const float*)d_in[18];
    const float* b2      = (const float*)d_in[19];
    const float* lnf_w   = (const float*)d_in[20];
    const float* lnf_b   = (const float*)d_in[21];
    const float* w_out   = (const float*)d_in[22];
    const float* b_out   = (const float*)d_in[23];
    float* out = (float*)d_out;

    const int M = B_ * S_;  // 4096 token rows
    char* ws = (char*)d_ws;
    size_t off = 0;
    auto take = [&](size_t bytes) -> char* {
        char* p = ws + off;
        off += (bytes + 255) & ~(size_t)255;
        return p;
    };

    float* X  = (float*)take((size_t)M * D_ * 4);        // fp32 residual stream
    bf16* Hb  = (bf16*)take((size_t)M * D_ * 2);         // LN output (GEMM A)
    bf16* Qb  = (bf16*)take((size_t)M * D_ * 2);
    bf16* Kb  = (bf16*)take((size_t)M * D_ * 2);
    bf16* Vb  = (bf16*)take((size_t)M * D_ * 2);
    bf16* Ob  = (bf16*)take((size_t)M * D_ * 2);
    bf16* GV  = (bf16*)take((size_t)M * 2 * FF_ * 2);
    bf16* FFb = (bf16*)take((size_t)M * FF_ * 2);

    // Grids: x = M tiles (fast, shares one weight panel in L2), y = N tiles
    const dim3 gDD(M / 128, D_ / 128);
    const dim3 gFF(M / 128, 2 * FF_ / 128);
    const dim3 gVV(M / 128, V_ / 128);

    // x = LN(emb[tokens])
    embed_ln_kernel<<<M, 256, 0, stream>>>(tokens, tok_emb, ln_e_w, ln_e_b, X);

    for (int l = 0; l < L_; ++l) {
        const size_t oDD = (size_t)l * D_ * D_;

        // ---- attention block
        ln_bf16_kernel<<<M, 256, 0, stream>>>(X, ln1_w + l * D_, ln1_b + l * D_, Hb);

        gemm_bf16_kernel<1><<<gDD, 256, 0, stream>>>(Hb, wq + oDD, bq + l * D_, nullptr, Qb, M, D_, D_);
        gemm_bf16_kernel<1><<<gDD, 256, 0, stream>>>(Hb, wk + oDD, bk + l * D_, nullptr, Kb, M, D_, D_);
        gemm_bf16_kernel<1><<<gDD, 256, 0, stream>>>(Hb, wvw + oDD, bv + l * D_, nullptr, Vb, M, D_, D_);

        flash_attn_kernel<<<B_ * H_ * (S_ / 64), 128, 0, stream>>>(Qb, Kb, Vb, Ob);

        gemm_bf16_kernel<0><<<gDD, 256, 0, stream>>>(Ob, wo + oDD, bo + l * D_, X, X, M, D_, D_);

        // ---- GEGLU MLP block
        ln_bf16_kernel<<<M, 256, 0, stream>>>(X, ln2_w + l * D_, ln2_b + l * D_, Hb);

        gemm_bf16_kernel<1><<<gFF, 256, 0, stream>>>(Hb, w1 + (size_t)l * D_ * 2 * FF_,
                                                     b1 + l * 2 * FF_, nullptr, GV, M, 2 * FF_, D_);
        geglu_kernel<<<4096, 256, 0, stream>>>(GV, FFb, M);

        gemm_bf16_kernel<0><<<gDD, 256, 0, stream>>>(FFb, w2 + (size_t)l * FF_ * D_,
                                                     b2 + l * D_, X, X, M, D_, FF_);
    }

    // final LN + LM head
    ln_bf16_kernel<<<M, 256, 0, stream>>>(X, lnf_w, lnf_b, Hb);
    gemm_bf16_kernel<0><<<gVV, 256, 0, stream>>>(Hb, w_out, b_out, nullptr, out, M, V_, D_);
}
